// FactorNetCombineLN220906_75703093559374
// MI455X (gfx1250) — matrix operand order repaired and checked by reading, not compiled, not run on hardware
//
#include <hip/hip_runtime.h>
#include <hip/hip_bf16.h>
#include <math.h>

typedef __attribute__((ext_vector_type(8)))  float  v8f;
typedef __attribute__((ext_vector_type(8)))  __bf16 v8bf;
typedef __attribute__((ext_vector_type(16))) __bf16 v16bf;

#if __has_builtin(__builtin_amdgcn_global_load_async_to_lds_b128) && __has_builtin(__builtin_amdgcn_s_wait_asynccnt)
#define USE_ASYNC_LDS 1
typedef int vint4 __attribute__((vector_size(16)));
#else
#define USE_ASYNC_LDS 0
#endif

#define B_BATCH 8192
#define C_DIM   64
#define NDAYS   10
#define NPAIR   2016
#define CH1     4800
#define M_ROWS  (B_BATCH * 3)   // 24576, multiple of 64

// ---------------- workspace layout (bytes) ----------------
#define A_ELEMS   ((size_t)M_ROWS * CH1)                 // 117,964,800
#define SA_BYTES  (A_ELEMS * 2)                          // 235,929,600
#define OFF_AHI   0ull
#define OFF_ALO   SA_BYTES
#define OFF_G     (2ull * SA_BYTES)
#define OFF_WHI   (3ull * SA_BYTES)
#define W_ELEMS   15300000ull
#define OFF_WLO   (OFF_WHI + W_ELEMS * 2)
#define W1_OFF 0ull
#define W2_OFF 11520000ull
#define W3_OFF 14400000ull
#define W4_OFF 15120000ull
#define ACT1_OFF 0ull
#define ACT2_OFF ((size_t)M_ROWS * 2400)
#define ACT3_OFF ((size_t)M_ROWS * 3600)
#define ACT4_OFF ((size_t)M_ROWS * 4200)

// ---------------------------------------------------------------------------
__device__ __forceinline__ void bfsplit(float v, __bf16* __restrict__ hi,
                                        __bf16* __restrict__ lo, size_t i)
{
    __bf16 h = (__bf16)v;
    hi[i] = h;
    lo[i] = (__bf16)(v - (float)h);
}

__device__ __forceinline__ void ln3_write(__bf16* __restrict__ hi, __bf16* __restrict__ lo,
                                          size_t b, int col,
                                          float v0, float v1, float v2,
                                          const float* __restrict__ g,
                                          const float* __restrict__ bt)
{
    float m  = (v0 + v1 + v2) * (1.0f / 3.0f);
    float d0 = v0 - m, d1 = v1 - m, d2 = v2 - m;
    float r  = rsqrtf((d0 * d0 + d1 * d1 + d2 * d2) * (1.0f / 3.0f) + 1e-5f);
    size_t base = (b * 3) * (size_t)CH1 + col;
    bfsplit(d0 * r * g[0] + bt[0], hi, lo, base);
    bfsplit(d1 * r * g[1] + bt[1], hi, lo, base + CH1);
    bfsplit(d2 * r * g[2] + bt[2], hi, lo, base + 2 * CH1);
}

// ---------------------------------------------------------------------------
__global__ __launch_bounds__(256) void wcvt_kernel(const float* __restrict__ src,
                                                   __bf16* __restrict__ hi,
                                                   __bf16* __restrict__ lo, int n)
{
    int i = blockIdx.x * 256 + threadIdx.x;
    if (i < n) bfsplit(src[i], hi, lo, (size_t)i);
}

// ---------------------------------------------------------------------------
// Kernel 1: per-batch features + fused LN-over-w -> h2 (M x 4800, bf16 split)
// ---------------------------------------------------------------------------
__global__ __launch_bounds__(256) void feat_kernel(
    const float* __restrict__ x,
    const float* __restrict__ ln_ts_g, const float* __restrict__ ln_ts_b,
    const float* __restrict__ pe_w,    const float* __restrict__ pe_b,
    const float* __restrict__ pe_g,    const float* __restrict__ pe_beta,
    __bf16* __restrict__ ahi, __bf16* __restrict__ alo)
{
    __shared__ __align__(16) float xs[3 * NDAYS * C_DIM];
    __shared__ float mus[3 * C_DIM];
    __shared__ float stds[3 * C_DIM];

    int tid  = threadIdx.x;
    size_t b = blockIdx.x;
    const float* xb = x + b * (size_t)(30 * C_DIM);

    for (int i = tid; i < 1920; i += 256) xs[i] = xb[i];
    __syncthreads();

    if (tid < C_DIM) {
        int c = tid;
        float mu[3], ret[3], dec[3];
        for (int w = 0; w < 3; ++w) {
            float s = 0.f, ds = 0.f;
            for (int d = 0; d < NDAYS; ++d) {
                float v = xs[w * 640 + d * 64 + c];
                s += v; ds += v * ((float)(d + 1) * (1.0f / 55.0f));
            }
            mu[w] = s * 0.1f; dec[w] = ds;
            ret[w] = xs[w * 640 + 9 * 64 + c] / xs[w * 640 + c] - 1.0f;
            mus[w * 64 + c] = mu[w];
        }
        ln3_write(ahi, alo, b, 4160 + c, ret[0], ret[1], ret[2], ln_ts_g + 12, ln_ts_b + 12);
        ln3_write(ahi, alo, b, 4224 + c, dec[0], dec[1], dec[2], ln_ts_g + 15, ln_ts_b + 15);
        ln3_write(ahi, alo, b, 4288 + c, mu[0],  mu[1],  mu[2],  ln_ts_g + 18, ln_ts_b + 18);
    }
    __syncthreads();

    for (int o = tid; o < 448; o += 256) {
        const float* wrow = pe_w + (size_t)o * 640;
        float pv[3];
        for (int w = 0; w < 3; ++w) {
            float acc = 0.f;
            const float* xw0 = xs + w * 640;
            for (int c = 0; c < 64; ++c)
                for (int d = 0; d < 10; ++d)
                    acc += xw0[d * 64 + c] * wrow[c * 10 + d];
            pv[w] = acc + pe_b[o];
        }
        ln3_write(ahi, alo, b, 4352 + o, pv[0], pv[1], pv[2], pe_g, pe_beta);
    }
    __syncthreads();

    for (int i = tid; i < 1920; i += 256) {
        int w = i / 640, c = i & 63;
        xs[i] -= mus[w * 64 + c];
    }
    __syncthreads();

    if (tid < C_DIM) {
        int c = tid;
        float st[3], zs[3];
        for (int w = 0; w < 3; ++w) {
            float s = 0.f;
            for (int d = 0; d < NDAYS; ++d) {
                float v = xs[w * 640 + d * 64 + c]; s += v * v;
            }
            st[w] = sqrtf(s * (1.0f / 9.0f));
            stds[w * 64 + c] = st[w];
            zs[w] = mus[w * 64 + c] / st[w];
        }
        ln3_write(ahi, alo, b, 4032 + c, st[0], st[1], st[2], ln_ts_g + 6, ln_ts_b + 6);
        ln3_write(ahi, alo, b, 4096 + c, zs[0], zs[1], zs[2], ln_ts_g + 9, ln_ts_b + 9);
    }
    __syncthreads();

    for (int p = tid; p < NPAIR; p += 256) {
        int i = 0, rem = p, cnt = C_DIM - 1;
        while (rem >= cnt) { rem -= cnt; cnt--; i++; }
        int j = i + 1 + rem;
        float cv[3], cr[3];
        for (int w = 0; w < 3; ++w) {
            float s = 0.f;
            const float* xi = xs + w * 640 + i;
            const float* xj = xs + w * 640 + j;
            for (int d = 0; d < NDAYS; ++d) s += xi[d * 64] * xj[d * 64];
            cv[w] = s * (1.0f / 9.0f);
            cr[w] = cv[w] / (stds[w * 64 + i] * stds[w * 64 + j]);
        }
        ln3_write(ahi, alo, b, p,        cr[0], cr[1], cr[2], ln_ts_g,     ln_ts_b);
        ln3_write(ahi, alo, b, 2016 + p, cv[0], cv[1], cv[2], ln_ts_g + 3, ln_ts_b + 3);
    }
}

// ---------------------------------------------------------------------------
// Kernel 2: bf16x3 split-precision GEMM, v_wmma_f32_16x16x32_bf16,
// async-to-LDS staging when available, double-buffered LDS.
// ---------------------------------------------------------------------------
#define APAD 40   // LDS row stride in bf16 elems (80B)

__global__ __launch_bounds__(256) void gemm_bf16x3(
    const __bf16* __restrict__ Ahi, const __bf16* __restrict__ Alo,
    const __bf16* __restrict__ Whi, const __bf16* __restrict__ Wlo,
    const float* __restrict__ bias, float* __restrict__ Out,
    int N, int K)
{
    __shared__ __align__(16) __bf16 sAh[2][64 * APAD];
    __shared__ __align__(16) __bf16 sAl[2][64 * APAD];
    __shared__ __align__(16) __bf16 sBh[2][128 * APAD];
    __shared__ __align__(16) __bf16 sBl[2][128 * APAD];

    int tid  = threadIdx.x;
    int lane = tid & 31;
    int wid  = tid >> 5;
    int m0 = blockIdx.y * 64;
    int n0 = blockIdx.x * 128;
    int wm = (wid & 1) * 32;
    int wn = (wid >> 1) * 32;

    v8f acc[2][2] = {};

    int sr = tid >> 2;            // 0..63
    int sc = (tid & 3) * 8;       // 0,8,16,24
    uint4 z4; z4.x = z4.y = z4.z = z4.w = 0u;

    int lm  = lane & 15;
    int hb  = lane >> 4;
    int c0a = hb * 16;   // A: [c0a,c0a+16) contiguous
    int c0b = hb * 16;   // B: [c0b,c0b+16) contiguous

    auto compute = [&](int buf) {
        v16bf ah[2], al[2], bh[2], bl[2];
        #pragma unroll
        for (int i = 0; i < 2; ++i) {
            int r = (wm + i * 16 + lm) * APAD;
            v8bf p0 = *(const v8bf*)&sAh[buf][r + c0a];
            v8bf p1 = *(const v8bf*)&sAh[buf][r + c0a + 8];
            ah[i] = __builtin_shufflevector(p0, p1, 0,1,2,3,4,5,6,7,8,9,10,11,12,13,14,15);
            v8bf q0 = *(const v8bf*)&sAl[buf][r + c0a];
            v8bf q1 = *(const v8bf*)&sAl[buf][r + c0a + 8];
            al[i] = __builtin_shufflevector(q0, q1, 0,1,2,3,4,5,6,7,8,9,10,11,12,13,14,15);
        }
        #pragma unroll
        for (int j = 0; j < 2; ++j) {
            int r = (wn + j * 16 + lm) * APAD;
            v8bf p0 = *(const v8bf*)&sBh[buf][r + c0b];
            v8bf p1 = *(const v8bf*)&sBh[buf][r + c0b + 8];
            bh[j] = __builtin_shufflevector(p0, p1, 0,1,2,3,4,5,6,7,8,9,10,11,12,13,14,15);
            v8bf q0 = *(const v8bf*)&sBl[buf][r + c0b];
            v8bf q1 = *(const v8bf*)&sBl[buf][r + c0b + 8];
            bl[j] = __builtin_shufflevector(q0, q1, 0,1,2,3,4,5,6,7,8,9,10,11,12,13,14,15);
        }
        #pragma unroll
        for (int i = 0; i < 2; ++i)
            #pragma unroll
            for (int j = 0; j < 2; ++j) {
                acc[i][j] = __builtin_amdgcn_wmma_f32_16x16x32_bf16(false, ah[i], false, bh[j], (short)0, acc[i][j], false, false);
                acc[i][j] = __builtin_amdgcn_wmma_f32_16x16x32_bf16(false, ah[i], false, bl[j], (short)0, acc[i][j], false, false);
                acc[i][j] = __builtin_amdgcn_wmma_f32_16x16x32_bf16(false, al[i], false, bh[j], (short)0, acc[i][j], false, false);
            }
    };

    int nk = (K + 31) / 32;

#if USE_ASYNC_LDS
    // ---- async-to-LDS staging path (ASYNCcnt) ----
    auto issue = [&](int k0, int buf) {
        bool kin = (k0 + sc < K);
        unsigned aoff = sr * APAD + sc;
        unsigned boff = (sr + 64) * APAD + sc;
        if (kin) {
            size_t o = (size_t)(m0 + sr) * K + k0 + sc;
            __builtin_amdgcn_global_load_async_to_lds_b128(
                (vint4*)(Ahi + o), (vint4*)&sAh[buf][aoff], 0, 0);
            __builtin_amdgcn_global_load_async_to_lds_b128(
                (vint4*)(Alo + o), (vint4*)&sAl[buf][aoff], 0, 0);
        } else {
            *(uint4*)&sAh[buf][aoff] = z4;
            *(uint4*)&sAl[buf][aoff] = z4;
        }
        int gn0 = n0 + sr;
        if (gn0 < N && kin) {
            size_t o = (size_t)gn0 * K + k0 + sc;
            __builtin_amdgcn_global_load_async_to_lds_b128(
                (vint4*)(Whi + o), (vint4*)&sBh[buf][aoff], 0, 0);
            __builtin_amdgcn_global_load_async_to_lds_b128(
                (vint4*)(Wlo + o), (vint4*)&sBl[buf][aoff], 0, 0);
        } else {
            *(uint4*)&sBh[buf][aoff] = z4;
            *(uint4*)&sBl[buf][aoff] = z4;
        }
        int gn1 = n0 + sr + 64;
        if (gn1 < N && kin) {
            size_t o = (size_t)gn1 * K + k0 + sc;
            __builtin_amdgcn_global_load_async_to_lds_b128(
                (vint4*)(Whi + o), (vint4*)&sBh[buf][boff], 0, 0);
            __builtin_amdgcn_global_load_async_to_lds_b128(
                (vint4*)(Wlo + o), (vint4*)&sBl[buf][boff], 0, 0);
        } else {
            *(uint4*)&sBh[buf][boff] = z4;
            *(uint4*)&sBl[buf][boff] = z4;
        }
        // L2 prefetch two chunks ahead
        int kp = k0 + 64;
        if (kp + sc < K) {
            __builtin_prefetch(Ahi + (size_t)(m0 + sr) * K + kp + sc, 0, 1);
            if (gn0 < N) __builtin_prefetch(Whi + (size_t)gn0 * K + kp + sc, 0, 1);
        }
    };

    issue(0, 0);
    __builtin_amdgcn_s_wait_asynccnt(0);
    __syncthreads();
    int cur = 0;
    for (int ck = 1; ck < nk; ++ck) {
        issue(ck * 32, cur ^ 1);
        compute(cur);
        __builtin_amdgcn_s_wait_asynccnt(0);
        __syncthreads();
        cur ^= 1;
    }
    compute(cur);
#else
    // ---- fallback: register-staged path ----
    uint4 rah, ral, rbh0, rbl0, rbh1, rbl1;
    auto fetch = [&](int k0) {
        if (k0 + sc < K) {
            size_t o = (size_t)(m0 + sr) * K + k0 + sc;
            rah = *(const uint4*)(Ahi + o);
            ral = *(const uint4*)(Alo + o);
        } else { rah = z4; ral = z4; }
        int gn0 = n0 + sr;
        if (gn0 < N && k0 + sc < K) {
            size_t o = (size_t)gn0 * K + k0 + sc;
            rbh0 = *(const uint4*)(Whi + o);
            rbl0 = *(const uint4*)(Wlo + o);
        } else { rbh0 = z4; rbl0 = z4; }
        int gn1 = n0 + sr + 64;
        if (gn1 < N && k0 + sc < K) {
            size_t o = (size_t)gn1 * K + k0 + sc;
            rbh1 = *(const uint4*)(Whi + o);
            rbl1 = *(const uint4*)(Wlo + o);
        } else { rbh1 = z4; rbl1 = z4; }
    };
    auto stage = [&](int buf) {
        *(uint4*)&sAh[buf][sr * APAD + sc] = rah;
        *(uint4*)&sAl[buf][sr * APAD + sc] = ral;
        *(uint4*)&sBh[buf][sr * APAD + sc] = rbh0;
        *(uint4*)&sBl[buf][sr * APAD + sc] = rbl0;
        *(uint4*)&sBh[buf][(sr + 64) * APAD + sc] = rbh1;
        *(uint4*)&sBl[buf][(sr + 64) * APAD + sc] = rbl1;
    };
    fetch(0);
    stage(0);
    __syncthreads();
    int cur = 0;
    for (int ck = 1; ck < nk; ++ck) {
        fetch(ck * 32);
        compute(cur);
        stage(cur ^ 1);
        __syncthreads();
        cur ^= 1;
    }
    compute(cur);
#endif

    // epilogue
    #pragma unroll
    for (int i = 0; i < 2; ++i)
        #pragma unroll
        for (int j = 0; j < 2; ++j) {
            int col = n0 + wn + j * 16 + lm;
            if (col < N) {
                float bv = bias[col];
                #pragma unroll
                for (int v = 0; v < 8; ++v) {
                    int row = m0 + wm + i * 16 + hb * 8 + v;
                    Out[(size_t)row * N + col] = acc[i][j][v] + bv;
                }
            }
        }
}

// ---------------------------------------------------------------------------
// Kernel 3: LN over the 3-row group + ReLU, fp32 in -> bf16 split out
// ---------------------------------------------------------------------------
__global__ __launch_bounds__(256) void ln_relu_kernel(
    const float* __restrict__ g, __bf16* __restrict__ ahi, __bf16* __restrict__ alo,
    const float* __restrict__ gamma, const float* __restrict__ beta, int N)
{
    int idx = blockIdx.x * 256 + threadIdx.x;
    if (idx >= B_BATCH * N) return;
    int b = idx / N, o = idx - b * N;
    size_t base = (size_t)(3 * b) * N + o;
    float v0 = g[base], v1 = g[base + N], v2 = g[base + 2 * (size_t)N];
    float m  = (v0 + v1 + v2) * (1.0f / 3.0f);
    float d0 = v0 - m, d1 = v1 - m, d2 = v2 - m;
    float r  = rsqrtf((d0 * d0 + d1 * d1 + d2 * d2) * (1.0f / 3.0f) + 1e-5f);
    float o0 = d0 * r * gamma[0] + beta[0];
    float o1 = d1 * r * gamma[1] + beta[1];
    float o2 = d2 * r * gamma[2] + beta[2];
    bfsplit(o0 > 0.f ? o0 : 0.f, ahi, alo, base);
    bfsplit(o1 > 0.f ? o1 : 0.f, ahi, alo, base + N);
    bfsplit(o2 > 0.f ? o2 : 0.f, ahi, alo, base + 2 * (size_t)N);
}

// ---------------------------------------------------------------------------
// Kernel 4: 3-step LSTM (H=30) + LN + head; one wave32 per batch element,
// x_t staged in LDS once per step.
// ---------------------------------------------------------------------------
__global__ __launch_bounds__(256) void lstm_head(
    const __bf16* __restrict__ Xhi, const __bf16* __restrict__ Xlo,
    const float* __restrict__ w_ih, const float* __restrict__ w_hh,
    const float* __restrict__ b_ih, const float* __restrict__ b_hh,
    const float* __restrict__ ln2_g, const float* __restrict__ ln2_b,
    const float* __restrict__ lin_w, const float* __restrict__ lin_b,
    float* __restrict__ out)
{
    __shared__ float sh_h[8][32];
    __shared__ float sg[8][128];
    __shared__ float sh_x[8][304];
    int wid = threadIdx.x >> 5, lane = threadIdx.x & 31;
    int b = blockIdx.x * 8 + wid;

    if (lane < 30) sh_h[wid][lane] = 0.f;
    float c = 0.f;
    float ys0 = 0.f, ys1 = 0.f, ys2 = 0.f;
    __syncthreads();

    for (int t = 0; t < 3; ++t) {
        size_t xoff = (size_t)(3 * b + t) * 300;
        for (int o = lane; o < 300; o += 32)
            sh_x[wid][o] = (float)Xhi[xoff + o] + (float)Xlo[xoff + o];
        __syncthreads();
        #pragma unroll
        for (int q = 0; q < 4; ++q) {
            int gi = lane + q * 32;
            if (gi < 120) {
                float acc = b_ih[gi] + b_hh[gi];
                const float* wr = w_ih + (size_t)gi * 300;
                for (int o2 = 0; o2 < 300; ++o2) acc += wr[o2] * sh_x[wid][o2];
                const float* hr = w_hh + gi * 30;
                for (int j = 0; j < 30; ++j) acc += hr[j] * sh_h[wid][j];
                sg[wid][gi] = acc;
            }
        }
        __syncthreads();
        if (lane < 30) {
            float ig = sg[wid][lane],      fg = sg[wid][lane + 30];
            float gg = sg[wid][lane + 60], og = sg[wid][lane + 90];
            float si = 1.f / (1.f + expf(-ig));
            float sf = 1.f / (1.f + expf(-fg));
            float so = 1.f / (1.f + expf(-og));
            c = sf * c + si * tanhf(gg);
            float hn = so * tanhf(c);
            if (t == 0) ys0 = hn; else if (t == 1) ys1 = hn; else ys2 = hn;
            sh_h[wid][lane] = hn;
        }
        __syncthreads();
    }

    float partial = 0.f;
    if (lane < 30) {
        float m  = (ys0 + ys1 + ys2) * (1.0f / 3.0f);
        float d0 = ys0 - m, d1 = ys1 - m, d2 = ys2 - m;
        float r  = rsqrtf((d0 * d0 + d1 * d1 + d2 * d2) * (1.0f / 3.0f) + 1e-5f);
        float x0 = d0 * r * ln2_g[0] + ln2_b[0];
        float x1 = d1 * r * ln2_g[1] + ln2_b[1];
        float x2 = d2 * r * ln2_g[2] + ln2_b[2];
        partial = x0 * lin_w[lane * 3] + x1 * lin_w[lane * 3 + 1] + x2 * lin_w[lane * 3 + 2];
    }
    for (int off = 16; off > 0; off >>= 1) partial += __shfl_down(partial, off);
    if (lane == 0) out[b] = partial + lin_b[0];
}

// ---------------------------------------------------------------------------
extern "C" void kernel_launch(void* const* d_in, const int* in_sizes, int n_in,
                              void* d_out, int out_size, void* d_ws, size_t ws_size,
                              hipStream_t stream)
{
    (void)in_sizes; (void)n_in; (void)out_size; (void)ws_size;
    const float* x     = (const float*)d_in[0];
    const float* lntg  = (const float*)d_in[1];
    const float* lntb  = (const float*)d_in[2];
    const float* pe_w  = (const float*)d_in[3];
    const float* pe_b  = (const float*)d_in[4];
    const float* pe_g  = (const float*)d_in[5];
    const float* pe_bt = (const float*)d_in[6];
    const float* e1w = (const float*)d_in[7];  const float* e1b = (const float*)d_in[8];
    const float* e1g = (const float*)d_in[9];  const float* e1bb = (const float*)d_in[10];
    const float* e2w = (const float*)d_in[11]; const float* e2b = (const float*)d_in[12];
    const float* e2g = (const float*)d_in[13]; const float* e2bb = (const float*)d_in[14];
    const float* e3w = (const float*)d_in[15]; const float* e3b = (const float*)d_in[16];
    const float* e3g = (const float*)d_in[17]; const float* e3bb = (const float*)d_in[18];
    const float* e4w = (const float*)d_in[19]; const float* e4b = (const float*)d_in[20];
    const float* e4g = (const float*)d_in[21]; const float* e4bb = (const float*)d_in[22];
    const float* w_ih = (const float*)d_in[23]; const float* w_hh = (const float*)d_in[24];
    const float* b_ih = (const float*)d_in[25]; const float* b_hh = (const float*)d_in[26];
    const float* ln2g = (const float*)d_in[27]; const float* ln2b = (const float*)d_in[28];
    const float* linw = (const float*)d_in[29]; const float* linb = (const float*)d_in[30];

    char* ws = (char*)d_ws;
    __bf16* Ahi = (__bf16*)(ws + OFF_AHI);
    __bf16* Alo = (__bf16*)(ws + OFF_ALO);
    float*  G   = (float*)(ws + OFF_G);
    __bf16* Whi = (__bf16*)(ws + OFF_WHI);
    __bf16* Wlo = (__bf16*)(ws + OFF_WLO);
    float* out  = (float*)d_out;
    dim3 blk(256);

    wcvt_kernel<<<(11520000 + 255) / 256, blk, 0, stream>>>(e1w, Whi + W1_OFF, Wlo + W1_OFF, 11520000);
    wcvt_kernel<<<( 2880000 + 255) / 256, blk, 0, stream>>>(e2w, Whi + W2_OFF, Wlo + W2_OFF,  2880000);
    wcvt_kernel<<<(  720000 + 255) / 256, blk, 0, stream>>>(e3w, Whi + W3_OFF, Wlo + W3_OFF,   720000);
    wcvt_kernel<<<(  180000 + 255) / 256, blk, 0, stream>>>(e4w, Whi + W4_OFF, Wlo + W4_OFF,   180000);

    feat_kernel<<<B_BATCH, blk, 0, stream>>>(x, lntg, lntb, pe_w, pe_b, pe_g, pe_bt, Ahi, Alo);

    gemm_bf16x3<<<dim3((2400 + 127) / 128, M_ROWS / 64), blk, 0, stream>>>(
        Ahi, Alo, Whi + W1_OFF, Wlo + W1_OFF, e1b, G, 2400, 4800);
    ln_relu_kernel<<<(B_BATCH * 2400 + 255) / 256, blk, 0, stream>>>(
        G, Ahi + ACT1_OFF, Alo + ACT1_OFF, e1g, e1bb, 2400);

    gemm_bf16x3<<<dim3((1200 + 127) / 128, M_ROWS / 64), blk, 0, stream>>>(
        Ahi + ACT1_OFF, Alo + ACT1_OFF, Whi + W2_OFF, Wlo + W2_OFF, e2b, G, 1200, 2400);
    ln_relu_kernel<<<(B_BATCH * 1200 + 255) / 256, blk, 0, stream>>>(
        G, Ahi + ACT2_OFF, Alo + ACT2_OFF, e2g, e2bb, 1200);

    gemm_bf16x3<<<dim3((600 + 127) / 128, M_ROWS / 64), blk, 0, stream>>>(
        Ahi + ACT2_OFF, Alo + ACT2_OFF, Whi + W3_OFF, Wlo + W3_OFF, e3b, G, 600, 1200);
    ln_relu_kernel<<<(B_BATCH * 600 + 255) / 256, blk, 0, stream>>>(
        G, Ahi + ACT3_OFF, Alo + ACT3_OFF, e3g, e3bb, 600);

    gemm_bf16x3<<<dim3((300 + 127) / 128, M_ROWS / 64), blk, 0, stream>>>(
        Ahi + ACT3_OFF, Alo + ACT3_OFF, Whi + W4_OFF, Wlo + W4_OFF, e4b, G, 300, 600);
    ln_relu_kernel<<<(B_BATCH * 300 + 255) / 256, blk, 0, stream>>>(
        G, Ahi + ACT4_OFF, Alo + ACT4_OFF, e4g, e4bb, 300);

    lstm_head<<<B_BATCH / 8, blk, 0, stream>>>(Ahi + ACT4_OFF, Alo + ACT4_OFF,
                                               w_ih, w_hh, b_ih, b_hh,
                                               ln2g, ln2b, linw, linb, out);
}
